// UANBlock_48112223649897
// MI455X (gfx1250) — matrix operand order repaired and checked by reading, not compiled, not run on hardware
//
#include <hip/hip_runtime.h>
#include <hip/hip_bf16.h>

typedef __attribute__((ext_vector_type(16))) __bf16 v16bf;
typedef __attribute__((ext_vector_type(8)))  __bf16 v8bf;
typedef __attribute__((ext_vector_type(4)))  __bf16 v4bf;
typedef __attribute__((ext_vector_type(8)))  float  v8f;
typedef __attribute__((ext_vector_type(4)))  float  v4f;

#define NH   8
#define DKV  16
#define DIN  256
#define HD   128

// ---- LDS slab layout (bytes), one wave == one workgroup, 32 rows per wave ----
#define ROWB    528      // bf16 tile row stride: 256*2 + 16B pad (conflict-free b128)
#define QBF_OFF 0        // 32 x 528 = 16896
#define KBF_OFF 16896
#define VBF_OFF 33792
#define PF_OFF  50688    // f32 P=Q*V tile (32x256), row stride 1024 B -> 32768
#define VAT_OFF 83456    // bf16 V_att tile (32x128), row stride 272 B -> 8704
#define SLAB    92160    // 90 KB -> 3 workgroups per WGP (320 KB LDS)

// ---- prepped weight regions in d_ws (element = __bf16) ----
#define WQ_OFF  0
#define WK_OFF  32768
#define WV_OFF  65536
#define WO_OFF  98304
// each region: 64 fragments * 1024 B ([frag][e_hi][lane][8 elems])

// ---------------- weight prep: shuffle into WMMA B-fragment layout ----------------
// B 32x16 bf16 layout (ISA 7.12.2): lane<16 -> N=lane, K = 8*kg + e for e<8, 16 + 8*kg + (e-8) for e>=8
__global__ void uan_prep_weights(const float* __restrict__ Wq, const float* __restrict__ Wk,
                                 const float* __restrict__ Wv, const float* __restrict__ Wo,
                                 __bf16* __restrict__ wp) {
    int t = blockIdx.x * 256 + threadIdx.x;      // 0 .. 131071
    if (t >= 4 * 32768) return;
    int region = t >> 15;                        // 0..2 = Wq/Wk/Wv, 3 = Wo
    int idx    = t & 32767;
    int f      = idx >> 9;                       // fragment 0..63
    int rem    = idx & 511;
    int lane   = rem >> 4;
    int e      = rem & 15;
    int n      = lane & 15;
    int kg     = lane >> 4;
    int Klocal = (e >> 3) * 16 + kg * 8 + (e & 7); // B-matrix K index within chunk of 32
    float val;
    if (region < 3) {
        int h = f >> 3, kc = f & 7;
        int K = kc * 32 + Klocal;                // K in 0..255
        const float* W = (region == 0) ? Wq : (region == 1) ? Wk : Wv;
        val = W[K * HD + h * DKV + n];           // W is (256,128) row-major
    } else {
        int nt = f >> 2, kc = f & 3;
        int K = kc * 32 + Klocal;                // K in 0..127
        val = Wo[K * DIN + nt * DKV + n];        // W_o is (128,256) row-major
    }
    long o = (long)region * 32768 + f * 512 + ((e >> 3) << 8) + lane * 8 + (e & 7);
    wp[o] = (__bf16)val;
}

// ---------------- fragment loaders ----------------
// A 16x32 bf16 layout: lane<16 -> M=lane holds K = 8*kg + {0..7} and 16 + 8*kg + {0..7}
__device__ inline v16bf load_afrag(const char* tile, int rowb, int m, int kg, int kc) {
    const char* p = tile + m * rowb + kc * 64 + kg * 16;
    v8bf lo = *(const v8bf*)p;
    v8bf hi = *(const v8bf*)(p + 32);
    return __builtin_shufflevector(lo, hi, 0,1,2,3,4,5,6,7,8,9,10,11,12,13,14,15);
}

__device__ inline v16bf load_bfrag(const __bf16* frag, int lane) {
    const char* p = (const char*)frag + lane * 16;
    v8bf lo = *(const v8bf*)p;
    v8bf hi = *(const v8bf*)(p + 512);
    return __builtin_shufflevector(lo, hi, 0,1,2,3,4,5,6,7,8,9,10,11,12,13,14,15);
}

__device__ inline v8f bfwmma(v16bf a, v16bf b, v8f c) {
    return __builtin_amdgcn_wmma_f32_16x16x32_bf16(false, a, false, b, (short)0, c, false, false);
}

__device__ inline float redsum16(float v) {
    v += __shfl_xor(v, 8, 16);
    v += __shfl_xor(v, 4, 16);
    v += __shfl_xor(v, 2, 16);
    v += __shfl_xor(v, 1, 16);
    return v;
}
__device__ inline float redmax16(float v) {
    v = fmaxf(v, __shfl_xor(v, 8, 16));
    v = fmaxf(v, __shfl_xor(v, 4, 16));
    v = fmaxf(v, __shfl_xor(v, 2, 16));
    v = fmaxf(v, __shfl_xor(v, 1, 16));
    return v;
}
__device__ inline float sigmoidf_(float x) { return 1.0f / (1.0f + __expf(-x)); }

// ---------------- main fused kernel: one wave handles 32 batch rows ----------------
__launch_bounds__(32)
__global__ void uan_main(const float* __restrict__ Q, const float* __restrict__ K,
                         const float* __restrict__ V,
                         const float* __restrict__ Wl, const float* __restrict__ bl,
                         const float* __restrict__ Ws, const float* __restrict__ bs,
                         const __bf16* __restrict__ wp,
                         float* __restrict__ out) {
    extern __shared__ __attribute__((aligned(16))) char lds[];
    const int lane = threadIdx.x & 31;
    const int n  = lane & 15;     // N index / A row index
    const int kg = lane >> 4;     // lane-half selector
    const long rowbase = (long)blockIdx.x * 32;

    char* Qbf = lds + QBF_OFF;
    char* Kbf = lds + KBF_OFF;
    char* Vbf = lds + VBF_OFF;
    char* Pf  = lds + PF_OFF;
    char* Vat = lds + VAT_OFF;

    __builtin_prefetch(wp, 0, 0);   // weights are L2-hot; emits global_prefetch

    // ---- Phase 0: stage 32x256 tiles of Q/K/V -> LDS (bf16) and P=Q*V (f32) ----
    const float* Qg = Q + rowbase * DIN;
    const float* Kg = K + rowbase * DIN;
    const float* Vg = V + rowbase * DIN;
#pragma unroll 4
    for (int j = 0; j < 64; ++j) {
        int idx = j * 128 + lane * 4;       // coalesced float4 over the 32x256 tile
        int row = idx >> 8;
        int col = idx & 255;
        v4f q = *(const v4f*)(Qg + idx);
        v4f k = *(const v4f*)(Kg + idx);
        v4f v = *(const v4f*)(Vg + idx);
        *(v4bf*)(Qbf + row * ROWB + col * 2) = __builtin_convertvector(q, v4bf);
        *(v4bf*)(Kbf + row * ROWB + col * 2) = __builtin_convertvector(k, v4bf);
        *(v4bf*)(Vbf + row * ROWB + col * 2) = __builtin_convertvector(v, v4bf);
        *(v4f*)(Pf + row * 1024 + col * 4) = q * v;
    }

    const float wl0 = Wl[n], wl1 = Wl[16 + n], wsn = Ws[n];
    const float bl0 = bl[0], bl1 = bl[1], bs0 = bs[0];

    // ---- Phase 1: per-head-pair GEMMs; B frags shared across 2 M-tiles,
    //               A frags shared across 2 heads (halves L2 + LDS traffic) ----
    const __bf16* wq = wp + WQ_OFF;
    const __bf16* wk = wp + WK_OFF;
    const __bf16* wv = wp + WV_OFF;
#pragma unroll 1
    for (int hp = 0; hp < 4; ++hp) {
        v8f accq[2][2] = {{{}, {}}, {{}, {}}};   // [head-in-pair][m-tile]
        v8f acck[2][2] = {{{}, {}}, {{}, {}}};
        v8f accv[2][2] = {{{}, {}}, {{}, {}}};
#pragma unroll
        for (int kc = 0; kc < 8; ++kc) {
            int f0 = (2 * hp) * 8 + kc;
            int f1 = f0 + 8;
            v16bf bq0 = load_bfrag(wq + f0 * 512, lane);
            v16bf bk0 = load_bfrag(wk + f0 * 512, lane);
            v16bf bv0 = load_bfrag(wv + f0 * 512, lane);
            v16bf bq1 = load_bfrag(wq + f1 * 512, lane);
            v16bf bk1 = load_bfrag(wk + f1 * 512, lane);
            v16bf bv1 = load_bfrag(wv + f1 * 512, lane);
#pragma unroll
            for (int mt = 0; mt < 2; ++mt) {
                v16bf fa_q = load_afrag(Qbf + mt * 16 * ROWB, ROWB, n, kg, kc);
                v16bf fa_k = load_afrag(Kbf + mt * 16 * ROWB, ROWB, n, kg, kc);
                v16bf fa_v = load_afrag(Vbf + mt * 16 * ROWB, ROWB, n, kg, kc);
                accq[0][mt] = bfwmma(fa_q, bq0, accq[0][mt]);
                acck[0][mt] = bfwmma(fa_k, bk0, acck[0][mt]);
                accv[0][mt] = bfwmma(fa_v, bv0, accv[0][mt]);
                accq[1][mt] = bfwmma(fa_q, bq1, accq[1][mt]);
                acck[1][mt] = bfwmma(fa_k, bk1, acck[1][mt]);
                accv[1][mt] = bfwmma(fa_v, bv1, accv[1][mt]);
            }
        }
        // C layout: vgpr r, lanes 0-15: (M=r, N=lane); lanes 16-31: (M=8+r, N=lane-16)
#pragma unroll
        for (int hh = 0; hh < 2; ++hh) {
            int h = 2 * hp + hh;
#pragma unroll
            for (int mt = 0; mt < 2; ++mt) {
#pragma unroll
                for (int r = 0; r < 8; ++r) {
                    float left  = acck[hh][mt][r] * accq[hh][mt][r];   // Kh*Qh
                    float right = accv[hh][mt][r] * accq[hh][mt][r];   // Vh*Qh
                    float d0 = redsum16(left * wl0);                   // left @ Wl[0]
                    float d1 = redsum16(left * wl1);                   // left @ Wl[1]
                    float m0 = sigmoidf_(d0 + bl0);
                    float m1 = sigmoidf_(d1 + bl1);
                    float fx = left * (m0 * m1);                       // (Kh*M0)*(Qh*M1)
                    float mx = redmax16(fx);                           // softmax over 16
                    float ex = __expf(fx - mx);
                    float sm = redsum16(ex);
                    float sc = redsum16(ex * wsn) / sm + bs0;          // softmax @ Ws^T + bs
                    float va = sc * right;                             // (BH,1,1)@(BH,1,16)
                    int row = mt * 16 + r + 8 * kg;
                    *(__bf16*)(Vat + row * 272 + (h * DKV + n) * 2) = (__bf16)va;
                }
            }
        }
    }

    // ---- Phase 2: (32x128)@(128x256) W_o GEMM + forward gating + SiLU ----
    const __bf16* wo = wp + WO_OFF;
#pragma unroll 1
    for (int nt = 0; nt < 16; ++nt) {
        v8f acc0 = {}, acc1 = {};
#pragma unroll
        for (int kc = 0; kc < 4; ++kc) {
            v16bf fb  = load_bfrag(wo + (nt * 4 + kc) * 512, lane);    // shared by both M-tiles
            v16bf fa0 = load_afrag(Vat,             272, n, kg, kc);
            v16bf fa1 = load_afrag(Vat + 16 * 272,  272, n, kg, kc);
            acc0 = bfwmma(fa0, fb, acc0);
            acc1 = bfwmma(fa1, fb, acc1);
        }
#pragma unroll
        for (int r = 0; r < 8; ++r) {
            int row0 = r + 8 * kg;
            int row1 = 16 + row0;
            float p0 = *(const float*)(Pf + row0 * 1024 + (nt * DKV + n) * 4);
            float p1 = *(const float*)(Pf + row1 * 1024 + (nt * DKV + n) * 4);
            float a0 = p0 * acc0[r];
            float a1 = p1 * acc1[r];
            out[(rowbase + row0) * DIN + nt * DKV + n] = a0 * sigmoidf_(a0);
            out[(rowbase + row1) * DIN + nt * DKV + n] = a1 * sigmoidf_(a1);
        }
    }
}

extern "C" void kernel_launch(void* const* d_in, const int* in_sizes, int n_in,
                              void* d_out, int out_size, void* d_ws, size_t ws_size,
                              hipStream_t stream) {
    const float* Q   = (const float*)d_in[0];
    const float* K   = (const float*)d_in[1];
    const float* V   = (const float*)d_in[2];
    const float* Wq  = (const float*)d_in[3];
    const float* Wk  = (const float*)d_in[4];
    const float* Wv  = (const float*)d_in[5];
    const float* Wo  = (const float*)d_in[6];
    const float* Wl  = (const float*)d_in[7];
    const float* bl  = (const float*)d_in[8];
    const float* Ws  = (const float*)d_in[9];
    const float* bs  = (const float*)d_in[10];
    __bf16* wp = (__bf16*)d_ws;                    // 256 KB of prepped weights
    float* out = (float*)d_out;

    uan_prep_weights<<<512, 256, 0, stream>>>(Wq, Wk, Wv, Wo, wp);

    // 90 KB dynamic LDS per (single-wave) workgroup; CDNA5 allows up to 320 KB.
    (void)hipFuncSetAttribute((const void*)uan_main,
                              hipFuncAttributeMaxDynamicSharedMemorySize, SLAB);
    // 131072 rows / 32 rows-per-wave = 4096 single-wave workgroups
    uan_main<<<4096, 32, SLAB, stream>>>(Q, K, V, Wl, bl, Ws, bs, wp, out);
}
